// WarpNet_39857296507423
// MI455X (gfx1250) — matrix operand order repaired and checked by reading, not compiled, not run on hardware
//
#include <hip/hip_runtime.h>
#include <stdint.h>

// ---------------------------------------------------------------------------
// CDNA5 (gfx1250) WarpNet forward.
//  * 3x3 / 1x1 convs: implicit GEMM via v_wmma_f32_16x16x32_bf16 (wave32),
//    96-K (3 WMMA) per barrier, division-free incremental im2col gather.
//  * Cosine correlation: v_wmma_f32_16x16x32_f16, fused max/argmax (the
//    3136x3136 score matrix is never materialized).
// ---------------------------------------------------------------------------

typedef __attribute__((ext_vector_type(16))) __bf16     v16bf;
typedef __attribute__((ext_vector_type(16))) _Float16   v16h;
typedef __attribute__((ext_vector_type(8)))  float      v8f;

union BFrag { uint4 q[2]; v16bf v; };
union HFrag { uint4 q[2]; v16h  v; };

__device__ __forceinline__ unsigned short f2bf(float x) {
    union { float f; uint32_t u; } c; c.f = x;
    uint32_t u = c.u;
    uint32_t r = (u + 0x7FFFu + ((u >> 16) & 1u)) >> 16;
    if ((u & 0x7FFFFFFFu) > 0x7F800000u) r = (u >> 16) | 0x40u;  // quiet NaN
    return (unsigned short)r;
}

// ---------------------------------------------------------------------------
// Generic f32 -> bf16 conversion
// ---------------------------------------------------------------------------
__global__ void cvt_bf16_k(const float* __restrict__ in,
                           unsigned short* __restrict__ out, int n) {
    int i = blockIdx.x * blockDim.x + threadIdx.x;
    if (i < n) out[i] = f2bf(in[i]);
}

// ---------------------------------------------------------------------------
// 3x3 conv, implicit GEMM, bf16 WMMA.
//  block = 256 threads (8 waves). Block tile: 128 (Cout) x 16 (positions).
//  grid.x = B*Ho*Wo/16 (always exact here), grid.y = Cout/128.
//  pad==1 -> reflect pad 1;  pad==0 -> VALID.
//  K = Cin*9 is a multiple of 96 for every layer (Cin % 32 == 0), so the
//  main loop stages 96 K per barrier and issues 3 WMMAs per barrier.
// ---------------------------------------------------------------------------
__global__ __launch_bounds__(256)
void conv3x3_wmma(const unsigned short* __restrict__ xb,   // bf16 (B,Cin,Hin,Win)
                  const unsigned short* __restrict__ wb,   // bf16 (Cout, Cin*9)
                  const float* __restrict__ bias,
                  float* __restrict__ outF,                // (B,Cout,Ho,Wo) or null
                  unsigned short* __restrict__ outB,       // bf16, same shape, or null
                  int Bn, int Cin, int Hin, int Win,
                  int Cout, int Ho, int Wo, int stride, int pad) {
    const int K    = Cin * 9;
    const int HWin = Hin * Win;
    const int tid  = threadIdx.x;
    const int wv   = tid >> 5;
    const int lane = tid & 31;
    const int HW   = Ho * Wo;
    const int n0   = blockIdx.x * 16;
    const int coBase = blockIdx.y * 128 + wv * 16;

    __shared__ alignas(16) unsigned short lb[2][1536];   // 2 x (3 chunks x 512)

    // ---- staging thread's fixed position; hoist reflect-pad offsets -------
    const int snl  = tid & 15;                 // position within the N tile
    const int skl0 = tid >> 4;                 // 0..15 ; handles skl0 + h*16
    int sb, ry0, ry1, ry2, cx0, cx1, cx2;
    {
        int n = n0 + snl;
        sb = n / HW; int rem = n - sb * HW;
        int soy = rem / Wo, sox = rem - soy * Wo;
        int yb = soy * stride - pad, xb0 = sox * stride - pad;
        int t;
        t = yb;     if (pad) { t = t < 0 ? -t : t; t = t >= Hin ? 2*Hin-2-t : t; } ry0 = t * Win;
        t = yb + 1; if (pad) { t = t < 0 ? -t : t; t = t >= Hin ? 2*Hin-2-t : t; } ry1 = t * Win;
        t = yb + 2; if (pad) { t = t < 0 ? -t : t; t = t >= Hin ? 2*Hin-2-t : t; } ry2 = t * Win;
        t = xb0;     if (pad) { t = t < 0 ? -t : t; t = t >= Win ? 2*Win-2-t : t; } cx0 = t;
        t = xb0 + 1; if (pad) { t = t < 0 ? -t : t; t = t >= Win ? 2*Win-2-t : t; } cx1 = t;
        t = xb0 + 2; if (pad) { t = t < 0 ? -t : t; t = t >= Win ? 2*Win-2-t : t; } cx2 = t;
    }
    const unsigned short* xbase = xb + (size_t)sb * Cin * HWin;

    // ---- A fragment addressing (weight rows are contiguous in K) ----------
    const int am  = lane & 15;
    const int akb = (lane >= 16) ? 16 : 0;
    const unsigned short* wrow = wb + (size_t)(coBase + am) * K;

    v8f acc = {0.f,0.f,0.f,0.f,0.f,0.f,0.f,0.f};

    // incremental (ci, r) for Kidx = kc + skl0 ; Kidx/9, Kidx%9 maintained
    int cc = skl0 / 9;
    int rr = skl0 - cc * 9;
    for (int kc = 0, it = 0; kc < K; kc += 96, ++it) {
        unsigned short* buf = lb[it & 1];
        // stage 6 elements: Kidx = kc + skl0 + j*16, j = 0..5
        int c2 = cc, r2 = rr;
        #pragma unroll
        for (int j = 0; j < 6; ++j) {
            int ky  = (r2 >= 6) ? 2 : ((r2 >= 3) ? 1 : 0);
            int kx  = r2 - 3 * ky;
            int row = (ky == 0) ? ry0 : ((ky == 1) ? ry1 : ry2);
            int col = (kx == 0) ? cx0 : ((kx == 1) ? cx1 : cx2);
            unsigned short v = xbase[(size_t)c2 * HWin + row + col];
            buf[(j >> 1) * 512 + (snl + (j & 1) * 16) * 16 + skl0] = v;
            // advance Kidx by 16:  ci += 1 (+1 on wrap), r = (r + 7) mod 9
            r2 += 7; c2 += 1; if (r2 >= 9) { r2 -= 9; c2 += 1; }
        }
        // advance base Kidx by 96: ci += 10 (+1 on wrap), r = (r + 6) mod 9
        rr += 6; cc += 10; if (rr >= 9) { rr -= 9; cc += 1; }
        __syncthreads();
        #pragma unroll
        for (int c = 0; c < 3; ++c) {
            BFrag af, bf;
            const unsigned short* wp = wrow + kc + c * 32 + akb;
            af.q[0] = *(const uint4*)(wp);
            af.q[1] = *(const uint4*)(wp + 8);
            const uint4* lq = (const uint4*)(buf + c * 512 + lane * 16);
            bf.q[0] = lq[0];
            bf.q[1] = lq[1];
            acc = __builtin_amdgcn_wmma_f32_16x16x32_bf16(false, af.v, false, bf.v,
                                                          (short)0, acc, false, false);
        }
    }

    // ---- write out: C layout -> row m = i + 8*(lane>=16), col n = lane&15 --
    const int nl = lane & 15;
    int n = n0 + nl;
    int b = n / HW; int rem = n - b * HW;
    const int rsel = (lane >= 16) ? 8 : 0;
    #pragma unroll
    for (int i = 0; i < 8; ++i) {
        int co = coBase + i + rsel;
        float v = acc[i] + bias[co];
        size_t oi = ((size_t)b * Cout + co) * HW + rem;
        if (outF) outF[oi] = v;
        if (outB) outB[oi] = f2bf(v);
    }
}

// ---------------------------------------------------------------------------
// 1x1 conv (theta/phi head) on a 54x54 tensor, with the rpad(54->56) folded
// into the gather. Output theta[b][co][m], m over the 56x56 grid.
// K = 128: single staging pass, one barrier, 4 back-to-back WMMAs.
// grid.x = 4*56*56/16 = 784, block = 256 (8 waves == Cout 128).
// ---------------------------------------------------------------------------
__global__ __launch_bounds__(256)
void conv1x1_reflect56(const unsigned short* __restrict__ xb,  // bf16 (4,128,54,54)
                       const unsigned short* __restrict__ wb,  // bf16 (128,128)
                       const float* __restrict__ bias,
                       float* __restrict__ outF) {             // (4,128,3136)
    const int tid = threadIdx.x, wv = tid >> 5, lane = tid & 31;
    const int n0 = blockIdx.x * 16;
    const int coBase = wv * 16;
    __shared__ alignas(16) unsigned short lb[2048];            // 4 chunks x 512

    const int snl = tid & 15;
    const int skl0 = tid >> 4;
    size_t sbase;
    {
        int n = n0 + snl;
        int b = n / 3136; int p = n - b * 3136;
        int fy = p / 56, fx = p - fy * 56;
        int sy = fy - 1; sy = (sy < 0) ? -sy : sy; sy = (sy >= 54) ? (106 - sy) : sy;
        int sx = fx - 1; sx = (sx < 0) ? -sx : sx; sx = (sx >= 54) ? (106 - sx) : sx;
        sbase = ((size_t)b * 128) * 2916 + (size_t)sy * 54 + sx;
    }
    // stage all 128 K values (8 elements per thread), one barrier
    #pragma unroll
    for (int j = 0; j < 8; ++j) {
        int ci = (j >> 1) * 32 + (j & 1) * 16 + skl0;
        unsigned short v = xb[sbase + (size_t)ci * 2916];
        lb[(j >> 1) * 512 + (snl + (j & 1) * 16) * 16 + skl0] = v;
    }
    __syncthreads();

    const int am = lane & 15;
    const int akb = (lane >= 16) ? 16 : 0;
    const unsigned short* wrow = wb + (size_t)(coBase + am) * 128;

    v8f acc = {0.f,0.f,0.f,0.f,0.f,0.f,0.f,0.f};
    #pragma unroll
    for (int c = 0; c < 4; ++c) {
        BFrag af, bf;
        af.q[0] = *(const uint4*)(wrow + c * 32 + akb);
        af.q[1] = *(const uint4*)(wrow + c * 32 + akb + 8);
        const uint4* lq = (const uint4*)(lb + c * 512 + lane * 16);
        bf.q[0] = lq[0];
        bf.q[1] = lq[1];
        acc = __builtin_amdgcn_wmma_f32_16x16x32_bf16(false, af.v, false, bf.v,
                                                      (short)0, acc, false, false);
    }
    const int nl = lane & 15;
    int n = n0 + nl;
    int b = n / 3136; int p = n - b * 3136;
    const int rsel = (lane >= 16) ? 8 : 0;
    #pragma unroll
    for (int i = 0; i < 8; ++i) {
        int co = coBase + i + rsel;
        outF[((size_t)b * 128 + co) * 3136 + p] = acc[i] + bias[co];
    }
}

// ---------------------------------------------------------------------------
// InstanceNorm stats: one block per (b,c); stats[bc] = {mean, rstd}
// ---------------------------------------------------------------------------
__global__ __launch_bounds__(256)
void inorm_stats(const float* __restrict__ x, float* __restrict__ stats, int HW) {
    const int bc = blockIdx.x;
    const float* p = x + (size_t)bc * HW;
    float s = 0.f, ss = 0.f;
    for (int i = threadIdx.x; i < HW; i += 256) { float v = p[i]; s += v; ss += v * v; }
    __shared__ float rs[256], rq[256];
    rs[threadIdx.x] = s; rq[threadIdx.x] = ss;
    __syncthreads();
    for (int o = 128; o > 0; o >>= 1) {
        if (threadIdx.x < o) { rs[threadIdx.x] += rs[threadIdx.x + o];
                               rq[threadIdx.x] += rq[threadIdx.x + o]; }
        __syncthreads();
    }
    if (threadIdx.x == 0) {
        float mean = rs[0] / HW;
        float var  = rq[0] / HW - mean * mean;
        stats[bc * 2]     = mean;
        stats[bc * 2 + 1] = rsqrtf(var + 1e-5f);
    }
}

// inorm apply + optional residual add + PReLU; writes f32 and/or bf16 with a
// channel offset into a Ctot-channel destination (for the concat buffer).
__global__ void inorm_apply(const float* __restrict__ x, const float* __restrict__ stats,
                            const float* __restrict__ skip,
                            float* __restrict__ outF, unsigned short* __restrict__ outB,
                            int C, int HW, int coff, int Ctot, int total) {
    int i = blockIdx.x * blockDim.x + threadIdx.x;
    if (i >= total) return;
    int hw = i % HW;
    int bc = i / HW;
    int c = bc % C, b = bc / C;
    float v = (x[i] - stats[bc * 2]) * stats[bc * 2 + 1];
    size_t oi = ((size_t)b * Ctot + coff + c) * HW + hw;
    if (skip) v += skip[oi];
    v = (v >= 0.f) ? v : 0.25f * v;
    if (outF) outF[oi] = v;
    if (outB) outB[oi] = f2bf(v);
}

// ---------------------------------------------------------------------------
// Nearest x2 upsample on bf16
// ---------------------------------------------------------------------------
__global__ void upnearest2_bf16(const unsigned short* __restrict__ in,
                                unsigned short* __restrict__ out,
                                int BC, int H, int W) {
    int H2 = 2 * H, W2 = 2 * W;
    int total = BC * H2 * W2;
    int i = blockIdx.x * blockDim.x + threadIdx.x;
    if (i >= total) return;
    int x = i % W2; int t = i / W2; int y = t % H2; int bc = t / H2;
    out[i] = in[((size_t)bc * H + (y >> 1)) * W + (x >> 1)];
}

// Bilinear x2 (align_corners=False) into the concat buffer (f32 + bf16)
__global__ void bilinear2_cat(const float* __restrict__ in,
                              float* __restrict__ outF, unsigned short* __restrict__ outB,
                              int Bn, int C, int H, int W, int coff, int Ctot) {
    int H2 = 2 * H, W2 = 2 * W;
    int total = Bn * C * H2 * W2;
    int i = blockIdx.x * blockDim.x + threadIdx.x;
    if (i >= total) return;
    int x = i % W2; int t = i / W2; int y = t % H2;
    int c = (t / H2) % C; int b = t / (H2 * C);
    float sy = (y + 0.5f) * 0.5f - 0.5f;
    float sx = (x + 0.5f) * 0.5f - 0.5f;
    int y0 = (int)floorf(sy); float fy = sy - y0;
    int x0 = (int)floorf(sx); float fx = sx - x0;
    int y0c = max(y0, 0), y1c = min(y0 + 1, H - 1);
    int x0c = max(x0, 0), x1c = min(x0 + 1, W - 1);
    const float* p = in + ((size_t)b * C + c) * H * W;
    float v = p[y0c * W + x0c] * (1.f - fy) * (1.f - fx)
            + p[y0c * W + x1c] * (1.f - fy) * fx
            + p[y1c * W + x0c] * fy * (1.f - fx)
            + p[y1c * W + x1c] * fy * fx;
    size_t oi = ((size_t)b * Ctot + coff + c) * (size_t)(H2 * W2) + (size_t)y * W2 + x;
    outF[oi] = v;
    outB[oi] = f2bf(v);
}

// ---------------------------------------------------------------------------
// Center over positions (per (b,c)), in place. One block per (b,c).
// ---------------------------------------------------------------------------
__global__ __launch_bounds__(256)
void center_m(float* __restrict__ th, int M) {
    const int bc = blockIdx.x;
    float* p = th + (size_t)bc * M;
    float s = 0.f;
    for (int i = threadIdx.x; i < M; i += 256) s += p[i];
    __shared__ float rs[256];
    rs[threadIdx.x] = s;
    __syncthreads();
    for (int o = 128; o > 0; o >>= 1) {
        if (threadIdx.x < o) rs[threadIdx.x] += rs[threadIdx.x + o];
        __syncthreads();
    }
    float mean = rs[0] / M;
    for (int i = threadIdx.x; i < M; i += 256) p[i] -= mean;
}

// L2-normalize over channels per (b,m) and write transposed f16 [b][m][c]
__global__ void norm_to_f16(const float* __restrict__ th, _Float16* __restrict__ out,
                            int Bn, int C, int M) {
    int i = blockIdx.x * blockDim.x + threadIdx.x;
    if (i >= Bn * M) return;
    int b = i / M, m = i % M;
    const float* p = th + (size_t)b * C * M + m;
    float ss = 0.f;
    for (int c = 0; c < C; ++c) { float v = p[(size_t)c * M]; ss += v * v; }
    float inv = 1.f / (sqrtf(ss) + 2.220446049250313e-16f);
    _Float16* o = out + ((size_t)b * M + m) * C;
    for (int c = 0; c < C; ++c) o[c] = (_Float16)(p[(size_t)c * M] * inv);
}

// ---------------------------------------------------------------------------
// Correlation + fused max/argmax: f[b,m,n] = theta_m . phi_n  (K=128, f16 WMMA)
// grid = (196 m-tiles, B). block = 256 (8 waves); wave w covers n-tiles w,w+8,...
// ---------------------------------------------------------------------------
__global__ __launch_bounds__(256)
void corr_argmax(const _Float16* __restrict__ thT,  // (B,3136,128)
                 const _Float16* __restrict__ phT,  // (B,3136,128)
                 float* __restrict__ smap, int* __restrict__ amax) {
    const int b = blockIdx.y;
    const int mt = blockIdx.x;
    const int tid = threadIdx.x, wv = tid >> 5, lane = tid & 31;

    // A fragments (resident): rows m = mt*16 + (lane&15), K chunks of 32
    const int ml = lane & 15;
    const int akb = (lane >= 16) ? 16 : 0;
    const _Float16* arow = thT + ((size_t)b * 3136 + mt * 16 + ml) * 128;
    HFrag a[4];
    #pragma unroll
    for (int c = 0; c < 4; ++c) {
        a[c].q[0] = *(const uint4*)(arow + c * 32 + akb);
        a[c].q[1] = *(const uint4*)(arow + c * 32 + akb + 8);
    }

    float rmax[8]; int ridx[8];
    #pragma unroll
    for (int i = 0; i < 8; ++i) { rmax[i] = -3.0e38f; ridx[i] = 0; }

    const int nl  = lane & 15;
    const int bkb = (lane >= 16) ? 16 : 0;
    for (int nt = wv; nt < 196; nt += 8) {
        const _Float16* brow = phT + ((size_t)b * 3136 + nt * 16 + nl) * 128;
        v8f acc = {0.f,0.f,0.f,0.f,0.f,0.f,0.f,0.f};
        #pragma unroll
        for (int c = 0; c < 4; ++c) {
            HFrag bf;
            bf.q[0] = *(const uint4*)(brow + c * 32 + bkb);
            bf.q[1] = *(const uint4*)(brow + c * 32 + bkb + 8);
            acc = __builtin_amdgcn_wmma_f32_16x16x32_f16(false, a[c].v, false, bf.v,
                                                         (short)0, acc, false, false);
        }
        int ng = nt * 16 + nl;
        #pragma unroll
        for (int i = 0; i < 8; ++i) {
            float v = acc[i];
            if (v > rmax[i]) { rmax[i] = v; ridx[i] = ng; }  // strict >: keeps first
        }
    }

    __shared__ float sMax[16][128];
    __shared__ int   sIdx[16][128];
    const int col = wv * 16 + nl;
    const int rsel = (lane >= 16) ? 8 : 0;
    #pragma unroll
    for (int i = 0; i < 8; ++i) {
        sMax[i + rsel][col] = rmax[i];
        sIdx[i + rsel][col] = ridx[i];
    }
    __syncthreads();
    if (tid < 16) {
        float best = -3.0e38f; int bidx = 0x7fffffff;
        for (int c = 0; c < 128; ++c) {
            float v = sMax[tid][c]; int ix = sIdx[tid][c];
            if (v > best || (v == best && ix < bidx)) { best = v; bidx = ix; }
        }
        size_t o = (size_t)b * 3136 + mt * 16 + tid;
        smap[o] = best;
        amax[o] = bidx;
    }
}

// ---------------------------------------------------------------------------
// avg_pool2d(B_lab, 4): (B,3,224,224) -> (B,3,3136)
// ---------------------------------------------------------------------------
__global__ void avgpool4(const float* __restrict__ Blab, float* __restrict__ Bp, int Bn) {
    int total = Bn * 3 * 3136;
    int i = blockIdx.x * blockDim.x + threadIdx.x;
    if (i >= total) return;
    int p = i % 3136; int c = (i / 3136) % 3; int b = i / (3 * 3136);
    int fy = p / 56, fx = p % 56;
    const float* src = Blab + ((size_t)b * 3 + c) * 224 * 224;
    float s = 0.f;
    #pragma unroll
    for (int dy = 0; dy < 4; ++dy)
        #pragma unroll
        for (int dx = 0; dx < 4; ++dx)
            s += src[(fy * 4 + dy) * 224 + fx * 4 + dx];
    Bp[i] = s * (1.f / 16.f);
}

// Gather colors at argmax, then nearest x4 upsample of y and smap into d_out
__global__ void final_map(const float* __restrict__ Bp, const float* __restrict__ smap,
                          const int* __restrict__ amax, float* __restrict__ out, int Bn) {
    int total = Bn * 3136;
    int i = blockIdx.x * blockDim.x + threadIdx.x;
    if (i >= total) return;
    int b = i / 3136, p = i % 3136;
    int fy = p / 56, fx = p % 56;
    int idx = amax[i];
    float sv = smap[i];
    size_t yelem = (size_t)Bn * 3 * 224 * 224;
    for (int c = 0; c < 3; ++c) {
        float v = Bp[((size_t)b * 3 + c) * 3136 + idx];
        float* dst = out + ((size_t)b * 3 + c) * 224 * 224;
        for (int dy = 0; dy < 4; ++dy)
            for (int dx = 0; dx < 4; ++dx)
                dst[(fy * 4 + dy) * 224 + fx * 4 + dx] = v;
    }
    float* d2 = out + yelem + (size_t)b * 224 * 224;
    for (int dy = 0; dy < 4; ++dy)
        for (int dx = 0; dx < 4; ++dx)
            d2[(fy * 4 + dy) * 224 + fx * 4 + dx] = sv;
}

// ===========================================================================
// Host orchestration
// ===========================================================================
extern "C" void kernel_launch(void* const* d_in, const int* in_sizes, int n_in,
                              void* d_out, int out_size, void* d_ws, size_t ws_size,
                              hipStream_t stream) {
    (void)in_sizes; (void)n_in; (void)out_size; (void)ws_size;

    const float* B_lab = (const float*)d_in[0];

    // ---- workspace arena --------------------------------------------------
    char* wsb = (char*)d_ws;
    size_t off = 0;
    auto alloc = [&](size_t bytes) -> void* {
        void* p = wsb + off;
        off = (off + bytes + 255) & ~(size_t)255;
        return p;
    };

    auto cvt = [&](const float* src, unsigned short* dst, int n) {
        cvt_bf16_k<<<dim3((n + 255) / 256), dim3(256), 0, stream>>>(src, dst, n);
    };

    // ---- convert conv weights to bf16 ------------------------------------
    // order: w2a w2b w3a w3b w4a w4b w5a w5b r1w1 r1w2 r2w1 r2w2 wl1 wl2 wth wph
    const int widx[16] = {9,11,13,15,17,19,21,23,25,27,29,31,33,35,37,39};
    const int wsz[16]  = {110592,147456,221184,147456,884736,294912,1769472,294912,
                          2359296,2359296,2359296,2359296,1179648,294912,16384,16384};
    unsigned short* wbf[16];
    for (int i = 0; i < 16; ++i) {
        wbf[i] = (unsigned short*)alloc((size_t)wsz[i] * 2);
        cvt((const float*)d_in[widx[i]], wbf[i], wsz[i]);
    }
    const float* bia[16];
    for (int i = 0; i < 16; ++i) bia[i] = (const float*)d_in[widx[i] + 1];

    // ---- activation buffers (reused across the two branches) --------------
    const size_t BIG = 6422528;  // max(4*128*112*112, 4*512*56*56)
    unsigned short* XB2 = (unsigned short*)alloc(4816896ull * 2);
    unsigned short* XB3 = (unsigned short*)alloc(2408448ull * 2);
    unsigned short* XB4 = (unsigned short*)alloc(1204224ull * 2);
    unsigned short* XB5 = (unsigned short*)alloc( 602112ull * 2);
    float*          T0  = (float*)alloc(BIG * 4);
    float*          T1  = (float*)alloc(BIG * 4);
    unsigned short* A0  = (unsigned short*)alloc(BIG * 2);
    unsigned short* A1  = (unsigned short*)alloc(BIG * 2);
    float*          CATF = (float*)alloc(BIG * 4);            // (4,512,56,56)
    unsigned short* CATB = (unsigned short*)alloc(BIG * 2);
    float*          STATS = (float*)alloc(4ull * 512 * 2 * 4);
    _Float16*       THT  = (_Float16*)alloc(4ull * 3136 * 128 * 2);
    _Float16*       PHT  = (_Float16*)alloc(4ull * 3136 * 128 * 2);
    float*          SMAP = (float*)alloc(4ull * 3136 * 4);
    int*            IDX  = (int*)alloc(4ull * 3136 * 4);
    float*          BP   = (float*)alloc(4ull * 3 * 3136 * 4);

    auto conv = [&](const unsigned short* x, const unsigned short* w, const float* bias,
                    float* outF, unsigned short* outB,
                    int Cin, int Hin, int Win, int Cout, int stride, int pad) {
        int Ho = (Hin + 2 * pad - 3) / stride + 1;
        int Wo = (Win + 2 * pad - 3) / stride + 1;
        int ntiles = (4 * Ho * Wo) / 16;           // exact for all shapes used
        conv3x3_wmma<<<dim3(ntiles, Cout / 128), dim3(256), 0, stream>>>(
            x, w, bias, outF, outB, 4, Cin, Hin, Win, Cout, Ho, Wo, stride, pad);
    };
    auto norm = [&](const float* x, int C, int HW, const float* skip,
                    float* outF, unsigned short* outB, int coff, int Ctot) {
        inorm_stats<<<dim3(4 * C), dim3(256), 0, stream>>>(x, STATS, HW);
        int total = 4 * C * HW;
        inorm_apply<<<dim3((total + 255) / 256), dim3(256), 0, stream>>>(
            x, STATS, skip, outF, outB, C, HW, coff, Ctot, total);
    };

    // ---- one feature branch: inputs -> normalized transposed f16 head -----
    auto branch = [&](const float* x2, const float* x3, const float* x4, const float* x5,
                      const unsigned short* whead, const float* bhead, _Float16* posT) {
        cvt(x2, XB2, 4816896);
        cvt(x3, XB3, 2408448);
        cvt(x4, XB4, 1204224);
        cvt(x5, XB5,  602112);

        // f2: 96@112 -> 128@112 -> (stride 2) 128@56 -> cat ch 0
        conv(XB2, wbf[0], bia[0], T0, nullptr, 96, 112, 112, 128, 1, 1);
        norm(T0, 128, 112 * 112, nullptr, nullptr, A0, 0, 128);
        conv(A0, wbf[1], bia[1], T0, nullptr, 128, 112, 112, 128, 2, 1);
        norm(T0, 128, 3136, nullptr, CATF, CATB, 0, 512);

        // f3: 192@56 -> 128@56 -> 128@56 -> cat ch 128
        conv(XB3, wbf[2], bia[2], T0, nullptr, 192, 56, 56, 128, 1, 1);
        norm(T0, 128, 3136, nullptr, nullptr, A0, 0, 128);
        conv(A0, wbf[3], bia[3], T0, nullptr, 128, 56, 56, 128, 1, 1);
        norm(T0, 128, 3136, nullptr, CATF, CATB, 128, 512);

        // f4: 384@28 -> 256@28 -> 128@28 -> bilinear x2 -> cat ch 256
        conv(XB4, wbf[4], bia[4], T0, nullptr, 384, 28, 28, 256, 1, 1);
        norm(T0, 256, 784, nullptr, nullptr, A0, 0, 256);
        conv(A0, wbf[5], bia[5], T0, nullptr, 256, 28, 28, 128, 1, 1);
        norm(T0, 128, 784, nullptr, T1, nullptr, 0, 128);
        {
            int tot = 4 * 128 * 56 * 56;
            bilinear2_cat<<<dim3((tot + 255) / 256), dim3(256), 0, stream>>>(
                T1, CATF, CATB, 4, 128, 28, 28, 256, 512);
        }

        // f5: 768@14 -> 256@14 -> nearest x2 -> 128@28 -> bilinear x2 -> cat ch 384
        conv(XB5, wbf[6], bia[6], T0, nullptr, 768, 14, 14, 256, 1, 1);
        norm(T0, 256, 196, nullptr, nullptr, A0, 0, 256);
        {
            int tot = 4 * 256 * 28 * 28;
            upnearest2_bf16<<<dim3((tot + 255) / 256), dim3(256), 0, stream>>>(
                A0, A1, 4 * 256, 14, 14);
        }
        conv(A1, wbf[7], bia[7], T0, nullptr, 256, 28, 28, 128, 1, 1);
        norm(T0, 128, 784, nullptr, T1, nullptr, 0, 128);
        {
            int tot = 4 * 128 * 56 * 56;
            bilinear2_cat<<<dim3((tot + 255) / 256), dim3(256), 0, stream>>>(
                T1, CATF, CATB, 4, 128, 28, 28, 384, 512);
        }

        // resblock 1 (skip = CATF, in-place update of CATF/CATB)
        conv(CATB, wbf[8], bia[8], T0, nullptr, 512, 56, 56, 512, 1, 1);
        norm(T0, 512, 3136, nullptr, nullptr, A0, 0, 512);
        conv(A0, wbf[9], bia[9], T1, nullptr, 512, 56, 56, 512, 1, 1);
        norm(T1, 512, 3136, CATF, CATF, CATB, 0, 512);

        // resblock 2
        conv(CATB, wbf[10], bia[10], T0, nullptr, 512, 56, 56, 512, 1, 1);
        norm(T0, 512, 3136, nullptr, nullptr, A0, 0, 512);
        conv(A0, wbf[11], bia[11], T1, nullptr, 512, 56, 56, 512, 1, 1);
        norm(T1, 512, 3136, CATF, CATF, CATB, 0, 512);

        // wl1 (VALID 56->54), wl2 (reflect on 54): bias only, no norm
        conv(CATB, wbf[12], bia[12], nullptr, A0, 512, 56, 56, 256, 1, 0);
        conv(A0,   wbf[13], bia[13], nullptr, A1, 256, 54, 54, 128, 1, 1);

        // 1x1 head with rpad(54->56) folded in; then center + L2-normalize
        conv1x1_reflect56<<<dim3(784), dim3(256), 0, stream>>>(A1, whead, bhead, T0);
        center_m<<<dim3(4 * 128), dim3(256), 0, stream>>>(T0, 3136);
        {
            int tot = 4 * 3136;
            norm_to_f16<<<dim3((tot + 255) / 256), dim3(256), 0, stream>>>(
                T0, posT, 4, 128, 3136);
        }
    };

    // branch A (theta head), branch B (phi head)
    branch((const float*)d_in[1], (const float*)d_in[2],
           (const float*)d_in[3], (const float*)d_in[4], wbf[14], bia[14], THT);
    branch((const float*)d_in[5], (const float*)d_in[6],
           (const float*)d_in[7], (const float*)d_in[8], wbf[15], bia[15], PHT);

    // correlation + WTA
    corr_argmax<<<dim3(196, 4), dim3(256), 0, stream>>>(THT, PHT, SMAP, IDX);

    // mapping_to_gray
    {
        int tot = 4 * 3 * 3136;
        avgpool4<<<dim3((tot + 255) / 256), dim3(256), 0, stream>>>(B_lab, BP, 4);
    }
    {
        int tot = 4 * 3136;
        final_map<<<dim3((tot + 255) / 256), dim3(256), 0, stream>>>(
            BP, SMAP, IDX, (float*)d_out, 4);
    }
}
